// NeRF_7078106103925
// MI455X (gfx1250) — matrix operand order repaired and checked by reading, not compiled, not run on hardware
//
#include <hip/hip_runtime.h>

// ---------------------------------------------------------------------------
// NeRF fused MLP for gfx1250 (MI455X): f16 WMMA, activations resident in LDS,
// weights pre-packed into WMMA B-fragment order, double-buffered LDS panels,
// async global->LDS staging (global_load_async_to_lds_b128 + s_wait_asynccnt).
// ---------------------------------------------------------------------------

typedef __attribute__((ext_vector_type(16))) _Float16 v16h;
typedef __attribute__((ext_vector_type(8)))  float    v8f;
typedef int v4i __attribute__((vector_size(16)));
typedef __attribute__((address_space(1))) v4i* gv4i_p;  // global
typedef __attribute__((address_space(3))) v4i* lv4i_p;  // LDS

// ---- geometry ----
#define MROWS 128           // samples per block
static constexpr int EP_STR = 72;    // f16 stride of ep buffer (63 used, pad->64)
static constexpr int EV_STR = 40;    // f16 stride of ev buffer (27 used, pad->32)
static constexpr int H_STR  = 264;   // f16 stride of activation buffer (256 used)
static constexpr int WBUF   = 8192;  // f16 per weight panel buffer (16 tiles * 512)

// packed weight layout offsets (f16 elements):
// m0 pts_w0 [64x256], m1..m4 pts_w1..4 [256x256], m5 pts_w5 [320x256],
// m6,m7 [256x256], m8 feat [256x256], m9 views [288x128]
__device__ constexpr int OFFS[11] = {
    0, 16384, 81920, 147456, 212992, 278528, 360448, 425984, 491520, 557056, 593920};
#define PACK_TOTAL 593920

// ---- optional async global->LDS path (probe via __has_builtin) ----
#if defined(__has_builtin)
#if __has_builtin(__builtin_amdgcn_global_load_async_to_lds_b128)
#define HAVE_ASYNC_LDS 1
#endif
#endif

__device__ __forceinline__ void cp16_g2l(const _Float16* g, _Float16* l) {
#ifdef HAVE_ASYNC_LDS
  __builtin_amdgcn_global_load_async_to_lds_b128((gv4i_p)g, (lv4i_p)l, 0, 0);
#else
  *(uint4*)l = *(const uint4*)g;
#endif
}

__device__ __forceinline__ void async_wait0() {
#ifdef HAVE_ASYNC_LDS
#if __has_builtin(__builtin_amdgcn_s_wait_asynccnt)
  __builtin_amdgcn_s_wait_asynccnt(0);
#else
  asm volatile("s_wait_asynccnt 0" ::: "memory");
#endif
#endif
}

// Stage one K-panel (NT 16x16-col tiles of a 32-K slab) global -> LDS.
template <int NT>
__device__ __forceinline__ void stage_panel(const _Float16* g, _Float16* l) {
  const int tid = threadIdx.x;
#pragma unroll
  for (int i = 0; i < (NT * 64) / 256; ++i) {
    const int c = tid + i * 256;
    cp16_g2l(g + c * 8, l + c * 8);
  }
}

// One dense layer: [128 x 32*KB] (f16, LDS) x [32*KB x 16*NT] (f16 packed, global)
// -> relu?(x@W + b) written back to dst (LDS, f16). A-source switches from
// (a0,a0str) to (a1,a1str) at splitKB (for the skip / views concatenations);
// a1off is the K-column where the a1 buffer starts.
template <int NT, bool RELU>
__device__ __forceinline__ void mlp_layer(
    const _Float16* __restrict__ wsrc, int KB,
    const _Float16* a0, int a0str, int splitKB,
    const _Float16* a1, int a1str, int a1off,
    const float* __restrict__ bias,
    _Float16* dst, _Float16* sw) {
  const int tid = threadIdx.x;
  const int lane = tid & 31;
  const int wave = tid >> 5;
  const int laneM = lane & 15;
  const bool hiHalf = lane >= 16;
  const int rowbase = wave * 16;  // each wave owns 16 rows

  v8f zero = {};
  v8f acc[NT];
#pragma unroll
  for (int nb = 0; nb < NT; ++nb) acc[nb] = zero;

  stage_panel<NT>(wsrc, sw);
  async_wait0();
  __syncthreads();

  for (int kb = 0; kb < KB; ++kb) {
    const int cur = kb & 1;
    if (kb + 1 < KB)
      stage_panel<NT>(wsrc + (size_t)(kb + 1) * NT * 512, sw + (cur ^ 1) * WBUF);

    // ---- A fragment (16x32 f16): per-lane runs k={0..15} / {16..31}
    const _Float16* ab;
    int astr, kcol;
    if (kb < splitKB) { ab = a0; astr = a0str; kcol = kb * 32; }
    else              { ab = a1; astr = a1str; kcol = kb * 32 - a1off; }
    const _Float16* ap = ab + (rowbase + laneM) * astr + kcol + (hiHalf ? 16 : 0);
    union { v16h v; struct { uint4 lo, hi; } q; } af;
    af.q.lo = *(const uint4*)(ap);
    af.q.hi = *(const uint4*)(ap + 8);

    const _Float16* wl = sw + cur * WBUF + lane * 16;
#pragma unroll
    for (int nb = 0; nb < NT; ++nb) {
      union { v16h v; struct { uint4 lo, hi; } q; } bf;
      const _Float16* bp = wl + nb * 512;
      bf.q.lo = *(const uint4*)(bp);
      bf.q.hi = *(const uint4*)(bp + 8);
      acc[nb] = __builtin_amdgcn_wmma_f32_16x16x32_f16(
          false, af.v, false, bf.v, (short)0, acc[nb], false, false);
    }
    async_wait0();
    __syncthreads();
  }

  // ---- epilogue: bias + relu + f16 store back into this wave's rows
#pragma unroll
  for (int nb = 0; nb < NT; ++nb) {
    const int n = nb * 16 + laneM;
    const float bv = bias[n];
#pragma unroll
    for (int g = 0; g < 8; ++g) {
      float v = acc[nb][g] + bv;
      if (RELU) v = fmaxf(v, 0.0f);
      const int m = rowbase + (hiHalf ? 8 : 0) + g;
      dst[m * H_STR + n] = (_Float16)v;
    }
  }
}

// ---------------------------------------------------------------------------
// Weight pre-pack: fp32 [K x N] -> f16 WMMA B-fragment order with K padding
// and concat row-remap.  Packed element order per tile: lane-major, 16 f16
// per lane, element j of lane L is (k = kb*32 + (L>=16?16:0) + j, n = nb*16 + L%16).
// ---------------------------------------------------------------------------
struct PackP {
  const float* w[10];
  _Float16* dst;
};

__global__ void pack_weights(PackP pp) {
  const int p = blockIdx.x * 256 + threadIdx.x;
  if (p >= PACK_TOTAL) return;
  int mi = 0;
#pragma unroll
  for (int i = 1; i < 10; ++i)
    if (p >= OFFS[i]) mi = i;
  const int local = p - OFFS[mi];
  const int N = (mi == 9) ? 128 : 256;
  const int NT = N >> 4;
  const int j = local & 15;
  const int lane = (local >> 4) & 31;
  const int tile = local >> 9;
  const int nb = tile % NT;
  const int kb = tile / NT;
  const int k = kb * 32 + ((lane & 16) ? 16 : 0) + j;
  const int n = nb * 16 + (lane & 15);
  int srck;
  if (mi == 0)      srck = (k < 63) ? k : -1;                         // ep pad
  else if (mi == 5) srck = (k < 63) ? k : ((k == 63) ? -1 : k - 1);   // [ep|h] skip
  else if (mi == 9) srck = (k < 283) ? k : -1;                        // [feat|ev]
  else              srck = k;
  const float v = (srck >= 0) ? pp.w[mi][srck * N + n] : 0.0f;
  pp.dst[p] = (_Float16)v;
}

// ---------------------------------------------------------------------------
// Fused NeRF kernel: 128 samples per block, 8 waves.
// ---------------------------------------------------------------------------
struct Params {
  const float* pts;
  const float* views;
  const float* pb[8];
  const float* views_b;
  const float* feat_b;
  const float* alpha_w;
  const float* alpha_b;
  const float* rgb_w;
  const float* rgb_b;
  const _Float16* wp;
  float* out;
};

__global__ __launch_bounds__(256) void nerf_main(Params P) {
  __shared__ _Float16 s_ep[MROWS * EP_STR];   // 18 KB
  __shared__ _Float16 s_ev[MROWS * EV_STR];   // 10 KB
  __shared__ _Float16 s_h[MROWS * H_STR];     // 66 KB
  __shared__ _Float16 s_w[2 * WBUF];          // 32 KB

  const int tid = threadIdx.x;
  const int base = blockIdx.x * MROWS;

  // ---- embeddings ----
  if (tid < MROWS) {
    const int s = base + tid;
    const float x0 = P.pts[3 * s + 0], x1 = P.pts[3 * s + 1], x2 = P.pts[3 * s + 2];
    _Float16* row = s_ep + tid * EP_STR;
    row[0] = (_Float16)x0; row[1] = (_Float16)x1; row[2] = (_Float16)x2;
    float f = 1.0f;
    for (int q = 0; q < 10; ++q) {
      row[3 + 6 * q + 0] = (_Float16)sinf(x0 * f);
      row[3 + 6 * q + 1] = (_Float16)sinf(x1 * f);
      row[3 + 6 * q + 2] = (_Float16)sinf(x2 * f);
      row[3 + 6 * q + 3] = (_Float16)cosf(x0 * f);
      row[3 + 6 * q + 4] = (_Float16)cosf(x1 * f);
      row[3 + 6 * q + 5] = (_Float16)cosf(x2 * f);
      f *= 2.0f;
    }
    row[63] = (_Float16)0.0f;
  } else {
    const int m = tid - MROWS;
    const int ray = (base + m) >> 6;  // N_SAMPLES = 64
    const float v0 = P.views[3 * ray + 0], v1 = P.views[3 * ray + 1],
                v2 = P.views[3 * ray + 2];
    _Float16* row = s_ev + m * EV_STR;
    row[0] = (_Float16)v0; row[1] = (_Float16)v1; row[2] = (_Float16)v2;
    float f = 1.0f;
    for (int q = 0; q < 4; ++q) {
      row[3 + 6 * q + 0] = (_Float16)sinf(v0 * f);
      row[3 + 6 * q + 1] = (_Float16)sinf(v1 * f);
      row[3 + 6 * q + 2] = (_Float16)sinf(v2 * f);
      row[3 + 6 * q + 3] = (_Float16)cosf(v0 * f);
      row[3 + 6 * q + 4] = (_Float16)cosf(v1 * f);
      row[3 + 6 * q + 5] = (_Float16)cosf(v2 * f);
      f *= 2.0f;
    }
    for (int c = 27; c < 32; ++c) row[c] = (_Float16)0.0f;
  }
  __syncthreads();

  const _Float16* wp = P.wp;
  // pts MLP
  mlp_layer<16, true>(wp + OFFS[0], 2, s_ep, EP_STR, 2, s_h, H_STR, 0, P.pb[0], s_h, s_w);
  mlp_layer<16, true>(wp + OFFS[1], 8, s_h, H_STR, 8, s_h, H_STR, 0, P.pb[1], s_h, s_w);
  mlp_layer<16, true>(wp + OFFS[2], 8, s_h, H_STR, 8, s_h, H_STR, 0, P.pb[2], s_h, s_w);
  mlp_layer<16, true>(wp + OFFS[3], 8, s_h, H_STR, 8, s_h, H_STR, 0, P.pb[3], s_h, s_w);
  mlp_layer<16, true>(wp + OFFS[4], 8, s_h, H_STR, 8, s_h, H_STR, 0, P.pb[4], s_h, s_w);
  // skip layer: input = [ep(64 padded) | h(256)]  -> K = 320
  mlp_layer<16, true>(wp + OFFS[5], 10, s_ep, EP_STR, 2, s_h, H_STR, 64, P.pb[5], s_h, s_w);
  mlp_layer<16, true>(wp + OFFS[6], 8, s_h, H_STR, 8, s_h, H_STR, 0, P.pb[6], s_h, s_w);
  mlp_layer<16, true>(wp + OFFS[7], 8, s_h, H_STR, 8, s_h, H_STR, 0, P.pb[7], s_h, s_w);
  __syncthreads();

  // alpha head (before feature overwrites h)
  float alpha = 0.0f;
  if (tid < MROWS) {
    const _Float16* hr = s_h + tid * H_STR;
    float a = P.alpha_b[0];
    for (int k = 0; k < 256; ++k) a += (float)hr[k] * P.alpha_w[k];
    alpha = a;
  }
  __syncthreads();

  // feature layer (no relu)
  mlp_layer<16, false>(wp + OFFS[8], 8, s_h, H_STR, 8, s_h, H_STR, 0, P.feat_b, s_h, s_w);
  // views layer: input = [feature(256) | ev(32 padded)] -> K = 288, N = 128
  mlp_layer<8, true>(wp + OFFS[9], 9, s_h, H_STR, 8, s_ev, EV_STR, 256, P.views_b, s_h, s_w);
  __syncthreads();

  // rgb head + output
  if (tid < MROWS) {
    const _Float16* hr = s_h + tid * H_STR;
    float r0 = P.rgb_b[0], r1 = P.rgb_b[1], r2 = P.rgb_b[2];
    for (int k = 0; k < 128; ++k) {
      const float hv = (float)hr[k];
      r0 += hv * P.rgb_w[3 * k + 0];
      r1 += hv * P.rgb_w[3 * k + 1];
      r2 += hv * P.rgb_w[3 * k + 2];
    }
    const int s = base + tid;
    P.out[4 * s + 0] = r0;
    P.out[4 * s + 1] = r1;
    P.out[4 * s + 2] = r2;
    P.out[4 * s + 3] = alpha;
  }
}

// ---------------------------------------------------------------------------
extern "C" void kernel_launch(void* const* d_in, const int* in_sizes, int n_in,
                              void* d_out, int out_size, void* d_ws, size_t ws_size,
                              hipStream_t stream) {
  (void)in_sizes; (void)n_in; (void)out_size; (void)ws_size;
  const float* pts = (const float*)d_in[0];
  const float* views = (const float*)d_in[1];
  const float* pw[8];
  const float* pb[8];
  for (int i = 0; i < 8; ++i) {
    pw[i] = (const float*)d_in[2 + 2 * i];
    pb[i] = (const float*)d_in[3 + 2 * i];
  }
  const float* views_w = (const float*)d_in[18];
  const float* views_b = (const float*)d_in[19];
  const float* feat_w = (const float*)d_in[20];
  const float* feat_b = (const float*)d_in[21];
  const float* alpha_w = (const float*)d_in[22];
  const float* alpha_b = (const float*)d_in[23];
  const float* rgb_w = (const float*)d_in[24];
  const float* rgb_b = (const float*)d_in[25];

  _Float16* wp = (_Float16*)d_ws;

  PackP pp;
  for (int i = 0; i < 8; ++i) pp.w[i] = pw[i];
  pp.w[8] = feat_w;
  pp.w[9] = views_w;
  pp.dst = wp;
  pack_weights<<<(PACK_TOTAL + 255) / 256, 256, 0, stream>>>(pp);

  Params P;
  P.pts = pts; P.views = views;
  for (int i = 0; i < 8; ++i) P.pb[i] = pb[i];
  P.views_b = views_b; P.feat_b = feat_b;
  P.alpha_w = alpha_w; P.alpha_b = alpha_b;
  P.rgb_w = rgb_w; P.rgb_b = rgb_b;
  P.wp = wp;
  P.out = (float*)d_out;
  nerf_main<<<(4096 * 64) / MROWS, 256, 0, stream>>>(P);
}
